// DynamicConv2d_10290741641791
// MI455X (gfx1250) — matrix operand order repaired and checked by reading, not compiled, not run on hardware
//
#include <hip/hip_runtime.h>
#include <hip/hip_bf16.h>

typedef _Float16 f16;
typedef __attribute__((ext_vector_type(16))) _Float16 v16h;
typedef __attribute__((ext_vector_type(8)))  _Float16 v8h;
typedef __attribute__((ext_vector_type(8)))  float    v8f;
typedef __attribute__((ext_vector_type(4)))  int      v4i;

#define C_IN   256
#define C_OUT  256
#define HH     56
#define WWID   56
#define HWN    (HH*WWID)      // 3136
#define NBASE  4
#define LDS_STRIDE 264        // 256 ci + 8 pad halves -> 132 dwords/row

#if defined(__has_builtin)
#if __has_builtin(__builtin_amdgcn_global_load_async_to_lds_b128)
#define HAVE_ASYNC_B128 1
#endif
#if __has_builtin(__builtin_amdgcn_s_wait_asynccnt)
#define HAVE_WAIT_ASYNC 1
#endif
#endif

typedef __attribute__((address_space(1))) v4i gv4i;   // global int4
typedef __attribute__((address_space(3))) v4i sv4i;   // LDS int4

__device__ __forceinline__ void async_copy16B(const f16* gsrc, f16* ldst) {
#ifdef HAVE_ASYNC_B128
    __builtin_amdgcn_global_load_async_to_lds_b128(
        (gv4i*)gsrc, (sv4i*)ldst, 0, 0);
#else
    unsigned int       lo = (unsigned int)(unsigned long long)ldst;   // LDS byte offset
    unsigned long long ga = (unsigned long long)gsrc;
    asm volatile("global_load_async_to_lds_b128 %0, %1, off"
                 :: "v"(lo), "v"(ga) : "memory");
#endif
}

__device__ __forceinline__ void wait_async0() {
#ifdef HAVE_WAIT_ASYNC
    __builtin_amdgcn_s_wait_asynccnt(0);
#else
    asm volatile("s_wait_asynccnt 0x0" ::: "memory");
#endif
}

// ---------------- Kernel 1: global average pool  pooled[b][c] ----------------
__global__ __launch_bounds__(128) void pool_kernel(const float* __restrict__ x,
                                                   float* __restrict__ pooled) {
    const int c = blockIdx.x, b = blockIdx.y, tid = threadIdx.x;
    const float* xp = x + ((size_t)b * C_IN + c) * HWN;
    float s = 0.f;
    for (int i = tid; i < HWN; i += 128) s += xp[i];
    __shared__ float red[128];
    red[tid] = s; __syncthreads();
    for (int off = 64; off > 0; off >>= 1) {
        if (tid < off) red[tid] += red[tid + off];
        __syncthreads();
    }
    if (tid == 0) pooled[b * C_IN + c] = red[0] * (1.0f / HWN);
}

// ---------------- Kernel 2: logits + softmax  attn[b][k] ----------------
__global__ __launch_bounds__(128) void attn_kernel(const float* __restrict__ pooled,
                                                   const float* __restrict__ attn_w,
                                                   const float* __restrict__ attn_b,
                                                   float* __restrict__ attn) {
    const int b = blockIdx.x, tid = threadIdx.x;
    const int k = tid >> 5, lane = tid & 31;
    const float* p = pooled + b * C_IN;
    const float* w = attn_w + k * C_IN;
    float s = 0.f;
    for (int c = lane; c < C_IN; c += 32) s += p[c] * w[c];
    for (int o = 16; o > 0; o >>= 1) s += __shfl_xor(s, o, 32);
    __shared__ float lg[NBASE];
    if (lane == 0) lg[k] = s + attn_b[k];
    __syncthreads();
    if (tid == 0) {
        float m = lg[0];
        for (int i = 1; i < NBASE; ++i) m = fmaxf(m, lg[i]);
        float e[NBASE], t = 0.f;
        for (int i = 0; i < NBASE; ++i) { e[i] = __expf(lg[i] - m); t += e[i]; }
        for (int i = 0; i < NBASE; ++i) attn[b * NBASE + i] = e[i] / t;
    }
}

// -------- Kernel 3: combined[b][j][cout][ci] (f16) = sum_k attn*W --------
__global__ __launch_bounds__(128) void combine_kernel(const float* __restrict__ bw,
                                                      const float* __restrict__ attn,
                                                      f16* __restrict__ comb) {
    const int b = blockIdx.y;
    const int idx = blockIdx.x * 128 + threadIdx.x;    // < 9*256*256
    const float a0 = attn[b*4+0], a1 = attn[b*4+1], a2 = attn[b*4+2], a3 = attn[b*4+3];
    const int j   = idx >> 16;
    const int rem = idx & 65535;                       // cout*256 + ci
    const size_t bstride = (size_t)C_OUT * C_IN * 9;
    const size_t woff = (size_t)rem * 9 + j;
    float s = a0 * bw[woff] + a1 * bw[bstride + woff]
            + a2 * bw[2*bstride + woff] + a3 * bw[3*bstride + woff];
    comb[(size_t)b * bstride + idx] = (f16)s;
}

// -------- Kernel 3b: transpose x to channel-last f16  xt[b][hw][ci] --------
__global__ __launch_bounds__(256) void xpose_kernel(const float* __restrict__ x,
                                                    f16* __restrict__ xt) {
    const int tid = threadIdx.x;
    const int hw0 = blockIdx.x * 64;
    const int b   = blockIdx.y;
    __shared__ f16 xs[64 * LDS_STRIDE];

    {   // coalesced read along hw, f32->f16, transpose into LDS
        const int fn = tid & 63, fci = tid >> 6;
        const float* xp = x + (size_t)b * C_IN * HWN + hw0 + fn;
        f16* dst = &xs[fn * LDS_STRIDE + fci];
#pragma unroll 4
        for (int it = 0; it < 64; ++it) {
            const int ci = it * 4 + fci;
            dst[it * 4] = (f16)xp[(size_t)ci * HWN];
        }
    }
    __syncthreads();
    {   // coalesced write along ci
        const int row = tid >> 2, q = tid & 3;
        const f16* srow = &xs[row * LDS_STRIDE + q * 64];
        f16* orow = xt + ((size_t)b * HWN + hw0 + row) * C_IN + q * 64;
#pragma unroll
        for (int i = 0; i < 8; ++i)
            *(v8h*)(orow + i * 8) = *(const v8h*)(srow + i * 8);
    }
}

// ---------------- Kernel 4: implicit-GEMM dynamic conv ----------------
// grid: (49 hw-tiles, 32 batch); block = 256 = 8 waves.
// Block tile: 256 cout x 64 hw. Wave tile: 32 cout x 64 hw (2x4 WMMA tiles).
__global__ __launch_bounds__(256) void dynconv_kernel(const f16* __restrict__ xt,
                                                      const f16* __restrict__ comb,
                                                      float* __restrict__ out) {
    const int tid = threadIdx.x;
    const int hw0 = blockIdx.x * 64;     // 49*64 == 3136, exact
    const int b   = blockIdx.y;

    __shared__ f16 xs[64 * LDS_STRIDE];  // xs[n][ci], 33.8 KB

    const int lane = tid & 31;
    const int wave = tid >> 5;           // wy = wave (0..7): cout = wave*32 + ...
    const int g  = lane >> 4;
    const int nl = lane & 15;

    // fill-phase mapping: thread covers row n = tid>>2, quarter q = tid&3 (128 B)
    const int fn  = tid >> 2;
    const int fq  = tid & 3;
    const int fhw = hw0 + fn;
    const int fh  = fhw / WWID, fw = fhw % WWID;

    size_t arow[2];
#pragma unroll
    for (int mt = 0; mt < 2; ++mt)
        arow[mt] = (size_t)(wave * 32 + mt * 16 + nl) * C_IN;

    v8f acc[2][4];
    const v8f vzero = {};
#pragma unroll
    for (int mt = 0; mt < 2; ++mt)
#pragma unroll
        for (int nt = 0; nt < 4; ++nt) acc[mt][nt] = vzero;

    const f16* xtb = xt + (size_t)b * HWN * C_IN;
    const v8h hzero = {};

    for (int j = 0; j < 9; ++j) {
        const int dh = j / 3 - 1, dw = j % 3 - 1;
        __syncthreads();
        {   // ---- async-copy shifted, zero-padded rows into LDS ----
            const int hh = fh + dh, ww = fw + dw;
            const bool valid = (hh >= 0) & (hh < HH) & (ww >= 0) & (ww < WWID);
            f16* dst = &xs[fn * LDS_STRIDE + fq * 64];
            if (valid) {
                const f16* src = xtb + (size_t)(fhw + dh * WWID + dw) * C_IN + fq * 64;
#pragma unroll
                for (int i = 0; i < 8; ++i)
                    async_copy16B(src + i * 8, dst + i * 8);
            } else {
#pragma unroll
                for (int i = 0; i < 8; ++i)
                    *(v8h*)(dst + i * 8) = hzero;
            }
        }
        wait_async0();
        __syncthreads();

        const f16* wj = comb + ((size_t)b * 9 + j) * (size_t)(C_OUT * C_IN);
#pragma unroll
        for (int kc = 0; kc < 8; ++kc) {
            const int ci0 = kc * 32;
            // A fragments: 16x32 f16 layout (halves 0-7: k=8g+0..7, 8-15: k=16+8g+0..7)
            v16h afr[2];
#pragma unroll
            for (int mt = 0; mt < 2; ++mt) {
                const f16* wr = wj + arow[mt] + ci0;
                v8h lo = *(const v8h*)(wr + 8 * g);
                v8h hi = *(const v8h*)(wr + 16 + 8 * g);
                afr[mt] = __builtin_shufflevector(lo, hi,
                          0,1,2,3,4,5,6,7,8,9,10,11,12,13,14,15);
            }
            // B fragments: 32x16 layout (halves 0-7: k=8g+0..7, 8-15: k=16+8g+0..7 of column nl)
            v16h bfr[4];
#pragma unroll
            for (int nt = 0; nt < 4; ++nt) {
                const f16* bp = &xs[(nt * 16 + nl) * LDS_STRIDE + ci0 + 8 * g];
                v8h lo = *(const v8h*)(bp);
                v8h hi = *(const v8h*)(bp + 16);
                bfr[nt] = __builtin_shufflevector(lo, hi,
                          0,1,2,3,4,5,6,7,8,9,10,11,12,13,14,15);
            }
#pragma unroll
            for (int mt = 0; mt < 2; ++mt)
#pragma unroll
                for (int nt = 0; nt < 4; ++nt)
                    acc[mt][nt] = __builtin_amdgcn_wmma_f32_16x16x32_f16(
                        false, afr[mt], false, bfr[nt],
                        (short)0, acc[mt][nt], false, false);
        }
    }

    // ---- store: C/D layout: VGPR r, lane l -> M = r + 8*(l/16), N = l%16
#pragma unroll
    for (int mt = 0; mt < 2; ++mt) {
        const int cm = wave * 32 + mt * 16 + 8 * g;
#pragma unroll
        for (int nt = 0; nt < 4; ++nt) {
            const int cn = hw0 + nt * 16 + nl;
            float* op = out + ((size_t)b * C_OUT + cm) * HWN + cn;
#pragma unroll
            for (int r = 0; r < 8; ++r) op[(size_t)r * HWN] = acc[mt][nt][r];
        }
    }
}

extern "C" void kernel_launch(void* const* d_in, const int* in_sizes, int n_in,
                              void* d_out, int out_size, void* d_ws, size_t ws_size,
                              hipStream_t stream) {
    const float* x  = (const float*)d_in[0];   // [32,256,56,56]
    const float* bw = (const float*)d_in[1];   // [4,256,256,3,3]
    const float* aw = (const float*)d_in[2];   // [4,256]
    const float* ab = (const float*)d_in[3];   // [4]
    float* out = (float*)d_out;                // [32,256,56,56]

    char* ws = (char*)d_ws;
    f16*   comb   = (f16*)ws;                                  // 37,748,736 B
    f16*   xt     = (f16*)(ws + 37748736);                     // 51,380,224 B
    float* pooled = (float*)(ws + 37748736 + 51380224);        // 32 KB
    float* attn   = (float*)(ws + 37748736 + 51380224 + 32768);// 512 B

    pool_kernel   <<<dim3(C_IN, 32), 128, 0, stream>>>(x, pooled);
    attn_kernel   <<<dim3(32),       128, 0, stream>>>(pooled, aw, ab, attn);
    combine_kernel<<<dim3(4608, 32), 128, 0, stream>>>(bw, attn, comb);
    xpose_kernel  <<<dim3(49, 32),   256, 0, stream>>>(x, xt);
    dynconv_kernel<<<dim3(49, 32),   256, 0, stream>>>(xt, comb, out);
}
